// TAPIR_83150566850830
// MI455X (gfx1250) — matrix operand order repaired and checked by reading, not compiled, not run on hardware
//
#include <hip/hip_runtime.h>

// ---------------------------------------------------------------- types
typedef __attribute__((ext_vector_type(16))) _Float16 v16h;
typedef __attribute__((ext_vector_type(8)))  _Float16 v8h;
typedef __attribute__((ext_vector_type(8)))  float    v8f;

// ---------------------------------------------------------------- WMMA helpers
// A: row-major [M][ldk] f16.  ISA 16-bit A 16x32 layout:
//  lanes 0-15 (M=lane):  elems 0..7 -> K=k0+0..7,  elems 8..15 -> K=k0+16..23
//  lanes 16-31(M=lane-16):elems 0..7 -> K=k0+8..15, elems 8..15 -> K=k0+24..31
__device__ inline v16h wmma_load_a(const _Float16* __restrict__ A, int ldk,
                                   int mbase, int k0, int lane) {
  int half = lane >> 4;
  int m = mbase + (lane & 15);
  const _Float16* p = A + (size_t)m * ldk + k0 + 8 * half;
  v8h lo = *(const v8h*)p;
  v8h hi = *(const v8h*)(p + 16);
  v16h r;
#pragma unroll
  for (int i = 0; i < 8; ++i) { r[i] = lo[i]; r[i + 8] = hi[i]; }
  return r;
}

// B: stored transposed, row-major [N][ldk] f16 (K contiguous).
//  lanes 0-15 (N=lane):   elems 0..7 -> K=k0+0..7,  elems 8..15 -> K=k0+16..23
//  lanes 16-31(N=lane-16): elems 0..7 -> K=k0+8..15, elems 8..15 -> K=k0+24..31
__device__ inline v16h wmma_load_b(const _Float16* __restrict__ BT, int ldk,
                                   int nbase, int k0, int lane) {
  int half = lane >> 4;
  int n = nbase + (lane & 15);
  const _Float16* p = BT + (size_t)n * ldk + k0 + 8 * half;
  v8h lo = *(const v8h*)p;
  v8h hi = *(const v8h*)(p + 16);
  v16h r;
#pragma unroll
  for (int i = 0; i < 8; ++i) { r[i] = lo[i]; r[i + 8] = hi[i]; }
  return r;
}

__device__ inline v8f wmma_f16(v16h a, v16h b, v8f c) {
  return __builtin_amdgcn_wmma_f32_16x16x32_f16(false, a, false, b, (short)0, c,
                                                false, false);
}

// ---------------------------------------------------------------- prep kernels
__global__ void k_convert_f16(const float* __restrict__ src,
                              _Float16* __restrict__ dst, int count) {
  int i = blockIdx.x * 256 + threadIdx.x;
  if (i < count) dst[i] = (_Float16)src[i];
}

__global__ void k_avgpool_fg(const float* __restrict__ fg,
                             float* __restrict__ fgavg) {
  int i = blockIdx.x * 256 + threadIdx.x;
  if (i >= 32 * 32 * 256) return;
  int c = i & 255, px = (i >> 8) & 31, py = i >> 13;
  int b0 = ((2 * py) * 64 + 2 * px) * 256 + c;
  fgavg[i] = 0.25f * (fg[b0] + fg[b0 + 256] + fg[b0 + 64 * 256] +
                      fg[b0 + 64 * 256 + 256]);
}

// mix_in_w [535][512] -> W1T [512][544] f16 (transposed, K padded 535->544)
__global__ void k_transpose_w1(const float* __restrict__ w,
                               _Float16* __restrict__ wt) {
  int i = blockIdx.x * 256 + threadIdx.x;
  if (i >= 512 * 544) return;
  int n = i / 544, k = i - n * 544;
  wt[i] = (k < 535) ? (_Float16)w[k * 512 + n] : (_Float16)0.f;
}

// mix_out_w [512][388] -> W2T [400][512] f16 (transposed, N padded 388->400)
__global__ void k_transpose_w2(const float* __restrict__ w,
                               _Float16* __restrict__ wt) {
  int i = blockIdx.x * 256 + threadIdx.x;
  if (i >= 400 * 512) return;
  int n = i / 512, k = i - n * 512;
  wt[i] = (n < 388) ? (_Float16)w[k * 388 + n] : (_Float16)0.f;
}

// feats = concat(hq, qf)  -> [1024][384] f32
__global__ void k_init_feats(const float* __restrict__ hq,
                             const float* __restrict__ qf,
                             float* __restrict__ feats) {
  int i = blockIdx.x * 256 + threadIdx.x;
  if (i >= 1024 * 384) return;
  int n = i / 384, c = i - n * 384;
  feats[i] = (c < 128) ? hq[n * 128 + c] : qf[n * 256 + (c - 128)];
}

// ---------------------------------------------------------------- cost volume
// cv[n][p] = sum_c qf[n][c]*fg[p][c]; M=1024,N=4096,K=256. One 16x16 tile/wave.
__global__ void k_cv_gemm(const _Float16* __restrict__ qfH,
                          const _Float16* __restrict__ fgH,
                          float* __restrict__ cv) {
  int wave = threadIdx.x >> 5, lane = threadIdx.x & 31;
  int tile = blockIdx.x * 8 + wave;          // 16384 tiles
  int tm = tile >> 8, tn = tile & 255;
  v8f acc = {};
#pragma unroll
  for (int k0 = 0; k0 < 256; k0 += 32) {
    v16h a = wmma_load_a(qfH, 256, tm * 16, k0, lane);
    v16h b = wmma_load_b(fgH, 256, tn * 16, k0, lane);
    acc = wmma_f16(a, b, acc);
  }
  int col = tn * 16 + (lane & 15);
  int rbase = tm * 16 + 8 * (lane >> 4);
#pragma unroll
  for (int r = 0; r < 8; ++r) cv[(size_t)(rbase + r) * 4096 + col] = acc[r];
}

// ---------------------------------------------------------------- stage B
// One block per query: conv1 -> occ_map(LDS f16) -> conv2+softargmax -> points,
// conv3(stride2)+mean -> tiny MLPs -> occ/expd.
__global__ __launch_bounds__(256) void k_stage_b(
    const float* __restrict__ cv, const float* __restrict__ w1,
    const float* __restrict__ b1, const float* __restrict__ w2,
    const float* __restrict__ b2, const float* __restrict__ w3,
    const float* __restrict__ b3, const float* __restrict__ w4,
    const float* __restrict__ b4, const float* __restrict__ ow,
    const float* __restrict__ ob, float* __restrict__ pos,
    float* __restrict__ occ_s, float* __restrict__ expd_s) {
  __shared__ float    s_cv[66 * 66];          // padded cv; reused for pmap
  __shared__ _Float16 s_occ[16 * 66 * 66];    // occ_map padded right/bottom
  __shared__ float    s_w3[32 * 16 * 9];
  __shared__ float    s_w1[144], s_b1[16], s_w2[144];
  __shared__ float    s_red[64];
  __shared__ float    s_part[32 * 8], s_o3[32], s_h4[16];

  int n = blockIdx.x, tid = threadIdx.x;
  for (int i = tid; i < 66 * 66; i += 256) s_cv[i] = 0.f;
  for (int i = tid; i < 16 * 66 * 66; i += 256) s_occ[i] = (_Float16)0.f;
  for (int i = tid; i < 144; i += 256) { s_w1[i] = w1[i]; s_w2[i] = w2[i]; }
  for (int i = tid; i < 4608; i += 256) s_w3[i] = w3[i];
  if (tid < 16) s_b1[tid] = b1[tid];
  __syncthreads();

  for (int i = tid; i < 4096; i += 256) {
    int y = i >> 6, x = i & 63;
    s_cv[(y + 1) * 66 + (x + 1)] = cv[(size_t)n * 4096 + i];
  }
  __syncthreads();

  // conv1 (1->16, SAME) + relu -> s_occ
  for (int i = tid; i < 16 * 4096; i += 256) {
    int oc = i >> 12, p = i & 4095, y = p >> 6, x = p & 63;
    const float* w = s_w1 + oc * 9;
    const float* s = s_cv + y * 66 + x;
    float a = s_b1[oc] + w[0] * s[0] + w[1] * s[1] + w[2] * s[2] +
              w[3] * s[66] + w[4] * s[67] + w[5] * s[68] + w[6] * s[132] +
              w[7] * s[133] + w[8] * s[134];
    s_occ[oc * 4356 + y * 66 + x] = (_Float16)fmaxf(a, 0.f);
  }
  __syncthreads();

  // conv2 (16->1, SAME) -> pmap (reuse s_cv)
  float b2v = b2[0];
  for (int p = tid; p < 4096; p += 256) {
    int y = p >> 6, x = p & 63;
    float a = b2v;
    for (int oc = 0; oc < 16; ++oc) {
      const _Float16* o = s_occ + oc * 4356;
      const float* w = s_w2 + oc * 9;
#pragma unroll
      for (int dy = 0; dy < 3; ++dy) {
        int yy = y + dy - 1;
        if (yy < 0) continue;                    // yy==64 reads padded zero
#pragma unroll
        for (int dx = 0; dx < 3; ++dx) {
          int xx = x + dx - 1;
          if (xx < 0) continue;
          a += w[dy * 3 + dx] * (float)o[yy * 66 + xx];
        }
      }
    }
    s_cv[p] = a;
  }
  __syncthreads();

  // softargmax (T=20): max, then sums of e, e*y, e*x
  float lmax = -1e30f;
  for (int p = tid; p < 4096; p += 256) lmax = fmaxf(lmax, s_cv[p] * 20.f);
#pragma unroll
  for (int o = 16; o >= 1; o >>= 1) lmax = fmaxf(lmax, __shfl_xor(lmax, o, 32));
  if ((tid & 31) == 0) s_red[tid >> 5] = lmax;
  __syncthreads();
  if (tid == 0) {
    float m = s_red[0];
    for (int i = 1; i < 8; ++i) m = fmaxf(m, s_red[i]);
    s_red[8] = m;
  }
  __syncthreads();
  float gmax = s_red[8];
  float ls = 0.f, lsy = 0.f, lsx = 0.f;
  for (int p = tid; p < 4096; p += 256) {
    float e = expf(s_cv[p] * 20.f - gmax);
    ls += e; lsy += e * (float)(p >> 6); lsx += e * (float)(p & 63);
  }
#pragma unroll
  for (int o = 16; o >= 1; o >>= 1) {
    ls += __shfl_xor(ls, o, 32);
    lsy += __shfl_xor(lsy, o, 32);
    lsx += __shfl_xor(lsx, o, 32);
  }
  if ((tid & 31) == 0) {
    s_red[16 + (tid >> 5)] = ls;
    s_red[24 + (tid >> 5)] = lsy;
    s_red[32 + (tid >> 5)] = lsx;
  }
  __syncthreads();
  if (tid == 0) {
    float S = 0, SY = 0, SX = 0;
    for (int i = 0; i < 8; ++i) { S += s_red[16 + i]; SY += s_red[24 + i]; SX += s_red[32 + i]; }
    pos[n * 2 + 0] = (SX / S) * 8.f;   // x * (512/64)
    pos[n * 2 + 1] = (SY / S) * 8.f;   // y * (512/64)
  }
  __syncthreads();

  // conv3 (16->32, stride2, pad(0,2) bottom/right) + relu + spatial mean
  int oc = tid >> 3, sub = tid & 7;
  float part = 0.f;
  for (int p = sub; p < 1024; p += 8) {
    int oy = p >> 5, ox = p & 31;
    float a = b3[oc];
    for (int ic = 0; ic < 16; ++ic) {
      const _Float16* o = s_occ + ic * 4356 + (2 * oy) * 66 + 2 * ox;
      const float* w = s_w3 + (oc * 16 + ic) * 9;
      a += w[0] * (float)o[0] + w[1] * (float)o[1] + w[2] * (float)o[2] +
           w[3] * (float)o[66] + w[4] * (float)o[67] + w[5] * (float)o[68] +
           w[6] * (float)o[132] + w[7] * (float)o[133] + w[8] * (float)o[134];
    }
    part += fmaxf(a, 0.f);
  }
  s_part[oc * 8 + sub] = part;
  __syncthreads();
  if (tid < 32) {
    float t = 0.f;
    for (int i = 0; i < 8; ++i) t += s_part[tid * 8 + i];
    s_o3[tid] = t * (1.f / 1024.f);
  }
  __syncthreads();
  if (tid < 16) {
    float a = b4[tid];
    for (int ic = 0; ic < 32; ++ic) a += s_o3[ic] * w4[ic * 16 + tid];
    s_h4[tid] = fmaxf(a, 0.f);
  }
  __syncthreads();
  if (tid < 2) {
    float a = ob[tid];
    for (int i = 0; i < 16; ++i) a += s_h4[i] * ow[i * 2 + tid];
    if (tid == 0) occ_s[n] = a; else expd_s[n] = a;
  }
}

// ---------------------------------------------------------------- per-iteration
// Bilinear 49-pt correlations over 3-level pyramid + assemble mlp_in (f16,[1024][544]).
__global__ __launch_bounds__(256) void k_corr_assemble(
    const float* __restrict__ hg, const float* __restrict__ fg,
    const float* __restrict__ fgavg, const float* __restrict__ pos,
    const float* __restrict__ occ_s, const float* __restrict__ expd_s,
    const float* __restrict__ feats, _Float16* __restrict__ mlpin) {
  __shared__ float s_q[384];
  __shared__ float s_pos[2];
  int n = blockIdx.x, tid = threadIdx.x;
  _Float16* row = mlpin + (size_t)n * 544;
  for (int c = tid; c < 384; c += 256) {
    float v = feats[(size_t)n * 384 + c];
    s_q[c] = v;
    row[4 + c] = (_Float16)v;
  }
  if (tid == 0) {
    row[0] = (_Float16)0.f; row[1] = (_Float16)0.f;
    row[2] = (_Float16)occ_s[n]; row[3] = (_Float16)expd_s[n];
    s_pos[0] = pos[n * 2]; s_pos[1] = pos[n * 2 + 1];
  }
  if (tid < 9) row[535 + tid] = (_Float16)0.f;   // K pad
  __syncthreads();

  int wave = tid >> 5, lane = tid & 31;
  float px = s_pos[0], py = s_pos[1];
  for (int l = 0; l < 3; ++l) {
    const float* g; int gh, gw, C, qo, co;
    if (l == 0)      { g = hg;    gh = 128; gw = 128; C = 128; qo = 0;   co = 388; }
    else if (l == 1) { g = fg;    gh = 64;  gw = 64;  C = 256; qo = 128; co = 437; }
    else             { g = fgavg; gh = 32;  gw = 32;  C = 256; qo = 128; co = 486; }
    float cy = py * ((float)gh / 512.f);
    float cx = px * ((float)gw / 512.f);
    for (int s = wave; s < 49; s += 8) {
      float sy = cy + (float)(s / 7 - 3);
      float sx = cx + (float)(s % 7 - 3);
      float y0f = floorf(sy), x0f = floorf(sx);
      float wy = sy - y0f, wx = sx - x0f;
      int y0 = min(max((int)y0f, 0), gh - 1);
      int y1 = min(max((int)y0f + 1, 0), gh - 1);
      int x0 = min(max((int)x0f, 0), gw - 1);
      int x1 = min(max((int)x0f + 1, 0), gw - 1);
      const float* r0 = g + (size_t)(y0 * gw) * C;
      const float* r1 = g + (size_t)(y1 * gw) * C;
      float w00 = (1.f - wy) * (1.f - wx), w01 = (1.f - wy) * wx;
      float w10 = wy * (1.f - wx), w11 = wy * wx;
      float acc = 0.f;
      for (int c = lane; c < C; c += 32) {
        float bi = w00 * r0[x0 * C + c] + w01 * r0[x1 * C + c] +
                   w10 * r1[x0 * C + c] + w11 * r1[x1 * C + c];
        acc += bi * s_q[qo + c];
      }
#pragma unroll
      for (int o = 16; o >= 1; o >>= 1) acc += __shfl_xor(acc, o, 32);
      if (lane == 0) row[co + s] = (_Float16)acc;
    }
  }
}

// GEMM1: [1024x544]x[544x512] + bias -> gelu -> f16 H
__global__ void k_gemm1(const _Float16* __restrict__ A,
                        const _Float16* __restrict__ BT,
                        const float* __restrict__ bias,
                        _Float16* __restrict__ H) {
  int wave = threadIdx.x >> 5, lane = threadIdx.x & 31;
  int tile = blockIdx.x * 8 + wave;          // 2048 tiles: 64 x 32
  int tm = tile >> 5, tn = tile & 31;
  v8f acc = {};
  for (int k0 = 0; k0 < 544; k0 += 32) {
    v16h a = wmma_load_a(A, 544, tm * 16, k0, lane);
    v16h b = wmma_load_b(BT, 544, tn * 16, k0, lane);
    acc = wmma_f16(a, b, acc);
  }
  int col = tn * 16 + (lane & 15);
  int rbase = tm * 16 + 8 * (lane >> 4);
  float bb = bias[col];
#pragma unroll
  for (int r = 0; r < 8; ++r) {
    float x = acc[r] + bb;
    float gelu = 0.5f * x *
                 (1.f + tanhf(0.7978845608f * (x + 0.044715f * x * x * x)));
    H[(size_t)(rbase + r) * 512 + col] = (_Float16)gelu;
  }
}

// GEMM2: [1024x512]x[512x388(+pad)] + bias, fused state update
__global__ void k_gemm2(const _Float16* __restrict__ H,
                        const _Float16* __restrict__ BT,
                        const float* __restrict__ bias,
                        float* __restrict__ pos, float* __restrict__ occ_s,
                        float* __restrict__ expd_s, float* __restrict__ feats) {
  int wave = threadIdx.x >> 5, lane = threadIdx.x & 31;
  int tile = blockIdx.x * 8 + wave;          // 1600 tiles: 64 x 25
  int tm = tile / 25, tn = tile % 25;
  v8f acc = {};
#pragma unroll
  for (int k0 = 0; k0 < 512; k0 += 32) {
    v16h a = wmma_load_a(H, 512, tm * 16, k0, lane);
    v16h b = wmma_load_b(BT, 512, tn * 16, k0, lane);
    acc = wmma_f16(a, b, acc);
  }
  int col = tn * 16 + (lane & 15);
  if (col >= 388) return;
  float bb = bias[col];
  int rbase = tm * 16 + 8 * (lane >> 4);
#pragma unroll
  for (int r = 0; r < 8; ++r) {
    float v = acc[r] + bb;
    int rowq = rbase + r;
    if (col == 0)       pos[rowq * 2 + 0] += v;
    else if (col == 1)  pos[rowq * 2 + 1] += v;
    else if (col == 2)  occ_s[rowq] += v;
    else if (col == 3)  expd_s[rowq] += v;
    else                feats[(size_t)rowq * 384 + (col - 4)] += v;
  }
}

__global__ void k_finalize(const float* __restrict__ pos,
                           const float* __restrict__ occ_s,
                           const float* __restrict__ expd_s,
                           float* __restrict__ out) {
  int n = blockIdx.x * 256 + threadIdx.x;
  if (n >= 1024) return;
  out[n * 4 + 0] = pos[n * 2 + 0];
  out[n * 4 + 1] = pos[n * 2 + 1];
  out[n * 4 + 2] = occ_s[n];
  out[n * 4 + 3] = expd_s[n];
}

// ---------------------------------------------------------------- host
extern "C" void kernel_launch(void* const* d_in, const int* in_sizes, int n_in,
                              void* d_out, int out_size, void* d_ws,
                              size_t ws_size, hipStream_t stream) {
  const float* fg  = (const float*)d_in[0];
  const float* hg  = (const float*)d_in[1];
  const float* qf  = (const float*)d_in[2];
  const float* hq  = (const float*)d_in[3];
  const float* w1  = (const float*)d_in[4];
  const float* b1  = (const float*)d_in[5];
  const float* w2  = (const float*)d_in[6];
  const float* b2  = (const float*)d_in[7];
  const float* w3  = (const float*)d_in[8];
  const float* b3  = (const float*)d_in[9];
  const float* w4  = (const float*)d_in[10];
  const float* b4  = (const float*)d_in[11];
  const float* ow  = (const float*)d_in[12];
  const float* ob  = (const float*)d_in[13];
  const float* miw = (const float*)d_in[14];
  const float* mib = (const float*)d_in[15];
  const float* mow = (const float*)d_in[16];
  const float* mob = (const float*)d_in[17];

  char* ws = (char*)d_ws;
  _Float16* fgH   = (_Float16*)(ws + 0);           // 2,097,152
  _Float16* qfH   = (_Float16*)(ws + 2097152);     //   524,288
  float*    fgavg = (float*)   (ws + 2621440);     // 1,048,576
  float*    cv    = (float*)   (ws + 3670016);     // 16,777,216
  _Float16* W1T   = (_Float16*)(ws + 20447232);    //   557,056
  _Float16* W2T   = (_Float16*)(ws + 21004288);    //   409,600
  _Float16* mlpin = (_Float16*)(ws + 21413888);    // 1,114,112
  _Float16* h16   = (_Float16*)(ws + 22528000);    // 1,048,576
  float*    pos   = (float*)   (ws + 23576576);    //     8,192
  float*    occ_s = (float*)   (ws + 23584768);    //     4,096
  float*    expd  = (float*)   (ws + 23588864);    //     4,096
  float*    feats = (float*)   (ws + 23592960);    // 1,572,864 -> 25,165,824 total

  // --- prep: f16 copies, pooled pyramid level, transposed f16 weights, feats
  k_convert_f16<<<4096, 256, 0, stream>>>(fg, fgH, 4096 * 256);
  k_convert_f16<<<1024, 256, 0, stream>>>(qf, qfH, 1024 * 256);
  k_avgpool_fg<<<1024, 256, 0, stream>>>(fg, fgavg);
  k_transpose_w1<<<1088, 256, 0, stream>>>(miw, W1T);
  k_transpose_w2<<<800, 256, 0, stream>>>(mow, W2T);
  k_init_feats<<<1536, 256, 0, stream>>>(hq, qf, feats);

  // --- cost volume GEMM (WMMA): 16384 tiles, 8 waves/block
  k_cv_gemm<<<2048, 256, 0, stream>>>(qfH, fgH, cv);

  // --- CNN head per query (LDS-resident)
  k_stage_b<<<1024, 256, 0, stream>>>(cv, w1, b1, w2, b2, w3, b3, w4, b4, ow,
                                      ob, pos, occ_s, expd);

  // --- 4 refinement iterations
  for (int it = 0; it < 4; ++it) {
    k_corr_assemble<<<1024, 256, 0, stream>>>(hg, fg, fgavg, pos, occ_s, expd,
                                              feats, mlpin);
    k_gemm1<<<256, 256, 0, stream>>>(mlpin, W1T, mib, h16);
    k_gemm2<<<200, 256, 0, stream>>>(h16, W2T, mob, pos, occ_s, expd, feats);
  }

  k_finalize<<<4, 256, 0, stream>>>(pos, occ_s, expd, (float*)d_out);
}
